// SSelfAttention_1082331758992
// MI455X (gfx1250) — matrix operand order repaired and checked by reading, not compiled, not run on hardware
//
#include <hip/hip_runtime.h>

// ---------------------------------------------------------------------------
// Node-axis multi-head self-attention for MI455X (gfx1250, wave32, WMMA).
// B=2, N=512 nodes, T=24, H=8 heads, d=8, E=64.
// 384 independent (b,t,h) attention problems of shape 512x512 with d=8.
// Strategy: f16 WMMA (V_WMMA_F32_16X16X32_F16) with f32 accumulation;
// energy/attn staged in LDS; softmax normalization folded into PV output.
// ---------------------------------------------------------------------------

#define NUM_HEADS 8
#define HEAD_DIM  8
#define EMBED     64
#define Bb        2
#define Nn        512
#define Tt        24
#define NPROB     (Bb * Tt * NUM_HEADS)   // 384 attention problems
#define QTILES    (Nn / 16)               // 32 query tiles per problem

typedef _Float16 v16h __attribute__((ext_vector_type(16)));
typedef _Float16 h8   __attribute__((ext_vector_type(8)));
typedef _Float16 h4   __attribute__((ext_vector_type(4)));
typedef float    v8f  __attribute__((ext_vector_type(8)));

union F16x16 { v16h v; h8 h2[2]; _Float16 s[16]; };
union HPack  { unsigned int u; _Float16 hh[2]; };

static __device__ __forceinline__ v8f wmma_f16(v16h a, v16h b, v8f c) {
  // 8 args: (neg_a, A, neg_b, B, c_mod, C, reuse_a, reuse_b)
  return __builtin_amdgcn_wmma_f32_16x16x32_f16(false, a, false, b,
                                                (short)0, c, false, false);
}

// ---------------------------------------------------------------------------
// Kernel 1: per-head input projections  q/k/v = x @ W^T  (f32 -> f16)
// Output layout: [p = (b*T + t)*H + h][n][d]  (contiguous N x 8 per problem)
// ---------------------------------------------------------------------------
__global__ __launch_bounds__(256) void proj_kernel(
    const float* __restrict__ values, const float* __restrict__ keys,
    const float* __restrict__ query,
    const float* __restrict__ Wv, const float* __restrict__ Wk,
    const float* __restrict__ Wq,
    _Float16* __restrict__ vp, _Float16* __restrict__ kp,
    _Float16* __restrict__ qp)
{
  __shared__ float sW[3][64];
  const int tid = threadIdx.x;
  if (tid < 64)        sW[0][tid]       = Wv[tid];
  else if (tid < 128)  sW[1][tid - 64]  = Wk[tid - 64];
  else if (tid < 192)  sW[2][tid - 128] = Wq[tid - 128];
  __syncthreads();

  const int idx = blockIdx.x * 256 + tid;        // flat (b,n,t,h)
  if (idx >= Bb * Nn * Tt * NUM_HEADS) return;
  const int h = idx & 7;
  const int t = (idx >> 3) % Tt;
  const int n = (idx / (8 * Tt)) % Nn;
  const int b = idx / (8 * Tt * Nn);

  const int src = idx * 8;                        // 8 contiguous f32 per head
  const int p   = (b * Tt + t) * NUM_HEADS + h;
  const int dst = p * (Nn * 8) + n * 8;

  float xv[8], xk[8], xq[8];
#pragma unroll
  for (int j = 0; j < 8; j++) {
    xv[j] = values[src + j]; xk[j] = keys[src + j]; xq[j] = query[src + j];
  }
  h8 ov, ok, oq;
#pragma unroll
  for (int i = 0; i < 8; i++) {
    float av = 0.f, ak = 0.f, aq = 0.f;
#pragma unroll
    for (int j = 0; j < 8; j++) {
      av += xv[j] * sW[0][i * 8 + j];
      ak += xk[j] * sW[1][i * 8 + j];
      aq += xq[j] * sW[2][i * 8 + j];
    }
    ov[i] = (_Float16)av; ok[i] = (_Float16)ak; oq[i] = (_Float16)aq;
  }
  *(h8*)(vp + dst) = ov;
  *(h8*)(kp + dst) = ok;
  *(h8*)(qp + dst) = oq;
}

// ---------------------------------------------------------------------------
// Kernel 2: attention. One wave32 per (problem, 16-row query tile).
//   energy tile = WMMA(Q 16x32 [K-dim padded 8->32 zeros], K^T 32x16)
//   softmax over 512 via LDS (two lanes per row, float4 passes),
//   normalization deferred into the PV output scaling
//   PV: 16 chained WMMA over K=512, V transposed in LDS for B-fragments
// Output o: f16, row-major [(b*N+n)*T + t][EMBED], ready for out-projection.
// ---------------------------------------------------------------------------
#define ES 516   // energy LDS stride (f32): 16B-aligned rows, bank-skewed
#define AS 520   // attn  LDS stride (f16): keeps 16B alignment
#define VS 528   // V^T   LDS stride (f16)

__global__ __launch_bounds__(32) void attn_kernel(
    const _Float16* __restrict__ qp, const _Float16* __restrict__ kp,
    const _Float16* __restrict__ vp, _Float16* __restrict__ o)
{
  __shared__ float    eLDS[16 * ES];    // 33.0 KB
  __shared__ _Float16 aLDS[16 * AS];    // 16.6 KB
  __shared__ _Float16 vtLDS[8 * VS];    //  8.4 KB   (total ~58 KB / block)

  const int lane  = threadIdx.x;
  const int p     = blockIdx.x >> 5;    // problem id
  const int qtile = blockIdx.x & 31;
  const int h     = p & 7;
  const int t     = (p >> 3) % Tt;
  const int b     = p / (8 * Tt);
  const int pbase = p * (Nn * 8);

  const int half = lane >> 4;           // 0: lanes 0-15, 1: lanes 16-31
  const int lc   = lane & 15;

  // --- stage V^T into LDS: vtLDS[c][k] = V[k][c]; two rows -> packed b32 ---
#pragma unroll 2
  for (int it = 0; it < 8; it++) {
    const int rr = it * 64 + lane * 2;
    h8 r0 = *(const h8*)(vp + pbase + rr * 8);
    h8 r1 = *(const h8*)(vp + pbase + rr * 8 + 8);
#pragma unroll
    for (int c = 0; c < 8; c++) {
      HPack pk; pk.hh[0] = r0[c]; pk.hh[1] = r1[c];
      *(unsigned int*)&vtLDS[c * VS + rr] = pk.u;
    }
  }

  // --- Q fragment: A 16x32 f16, K-dim 0..7 live, rest zero ---
  F16x16 aq = {};
  if (lane < 16)
    aq.h2[0] = *(const h8*)(qp + pbase + (qtile * 16 + lane) * 8);

  // --- energy = (Q K^T) / 8, 32 column tiles ---
  for (int j = 0; j < 32; j++) {
    F16x16 bk = {};
    if (lane < 16)
      bk.h2[0] = *(const h8*)(kp + pbase + (j * 16 + lane) * 8);
    v8f cz = {};
    v8f d = wmma_f16(aq.v, bk.v, cz);
#pragma unroll
    for (int r = 0; r < 8; r++)
      eLDS[(r + 8 * half) * ES + j * 16 + lc] = d[r] * 0.125f;
  }
  __syncthreads();

  // --- softmax: lanes (lc, lc+16) cooperate on row lc; float4 passes ---
  const int row   = lc;
  const int cbase = half * 256;
  float rmax = -3.0e38f;
  for (int c4 = 0; c4 < 256; c4 += 4) {
    float4 e4 = *(const float4*)&eLDS[row * ES + cbase + c4];
    rmax = fmaxf(rmax, fmaxf(fmaxf(e4.x, e4.y), fmaxf(e4.z, e4.w)));
  }
  rmax = fmaxf(rmax, __shfl_xor(rmax, 16, 32));
  float rsum = 0.f;
  for (int c4 = 0; c4 < 256; c4 += 4) {
    float4 e4 = *(const float4*)&eLDS[row * ES + cbase + c4];
    float x0 = __expf(e4.x - rmax);
    float x1 = __expf(e4.y - rmax);
    float x2 = __expf(e4.z - rmax);
    float x3 = __expf(e4.w - rmax);
    rsum += (x0 + x1) + (x2 + x3);
    h4 pk;
    pk[0] = (_Float16)x0; pk[1] = (_Float16)x1;
    pk[2] = (_Float16)x2; pk[3] = (_Float16)x3;
    *(h4*)&aLDS[row * AS + cbase + c4] = pk;      // unnormalized exp, b64
  }
  rsum += __shfl_xor(rsum, 16, 32);
  const float inv = 1.0f / rsum;                  // all 32 lanes: inv of row lc
  __syncthreads();

  // --- PV: out(16x8) = attn(16x512) @ V(512x8), K=512 in 16 WMMA steps ---
  const int off  = half * 8;    // A fragment K offset within 32-block
  const int colv = lane & 7;    // B columns 8..15 are don't-care -> replicate
  const int kb   = half * 8;    // B fragment K offset within 32-block
  v8f acc = {};
  for (int jj = 0; jj < 16; jj++) {
    F16x16 ua, ub;
    ua.h2[0] = *(const h8*)&aLDS[row * AS + jj * 32 + off];
    ua.h2[1] = *(const h8*)&aLDS[row * AS + jj * 32 + off + 16];
    ub.h2[0] = *(const h8*)&vtLDS[colv * VS + jj * 32 + kb];
    ub.h2[1] = *(const h8*)&vtLDS[colv * VS + jj * 32 + kb + 16];
    acc = wmma_f16(ua.v, ub.v, acc);
  }

  // --- fold in 1/rowsum, store f16 to o[(b*N+n)*T + t][h*8 + col] ---
  float sc[8];
#pragma unroll
  for (int r = 0; r < 8; r++)
    sc[r] = __shfl(inv, r + 8 * half, 32);        // inv lives in lanes 0..15
  if (lc < 8) {
#pragma unroll
    for (int r = 0; r < 8; r++) {
      const int nnode = qtile * 16 + r + 8 * half;
      const int mrow  = (b * Nn + nnode) * Tt + t;
      o[mrow * EMBED + h * 8 + lc] = (_Float16)(acc[r] * sc[r]);
    }
  }
}

// ---------------------------------------------------------------------------
// Kernel 3: out = o @ Wo^T + bo.  M = B*N*T = 24576, N = K = 64.
// One wave per 16x16 output tile; bias preloaded into the C accumulator.
// ---------------------------------------------------------------------------
__global__ __launch_bounds__(32) void outproj_kernel(
    const _Float16* __restrict__ o, const float* __restrict__ Wo,
    const float* __restrict__ bo, float* __restrict__ out)
{
  const int lane  = threadIdx.x;
  const int rtile = blockIdx.x >> 2;
  const int ctile = blockIdx.x & 3;
  const int lc    = lane & 15;
  const int half  = lane >> 4;
  const int n     = ctile * 16 + lc;          // output column

  const float bias = bo[n];
  v8f c;
#pragma unroll
  for (int i = 0; i < 8; i++) c[i] = bias;

  const int m   = rtile * 16 + lc;            // A row for this lane
  const int off = half * 8;
  const int kb  = half * 8;
#pragma unroll
  for (int kk = 0; kk < 2; kk++) {            // K = 64 in two 32-chunks
    F16x16 ua, ub;
    ua.h2[0] = *(const h8*)(o + m * EMBED + kk * 32 + off);
    ua.h2[1] = *(const h8*)(o + m * EMBED + kk * 32 + off + 16);
    const float* wrow = Wo + n * EMBED + kk * 32 + kb;  // B[k][n] = Wo[n][k]
#pragma unroll
    for (int e = 0; e < 16; e++) ub.s[e] = (_Float16)wrow[e < 8 ? e : e + 8];
    c = wmma_f16(ua.v, ub.v, c);
  }
#pragma unroll
  for (int r = 0; r < 8; r++) {
    const int mm = rtile * 16 + r + 8 * half;
    out[mm * EMBED + n] = c[r];
  }
}

// ---------------------------------------------------------------------------
extern "C" void kernel_launch(void* const* d_in, const int* in_sizes, int n_in,
                              void* d_out, int out_size, void* d_ws,
                              size_t ws_size, hipStream_t stream) {
  const float* values = (const float*)d_in[0];
  const float* keys   = (const float*)d_in[1];
  const float* query  = (const float*)d_in[2];
  const float* Wv     = (const float*)d_in[3];
  const float* Wk     = (const float*)d_in[4];
  const float* Wq     = (const float*)d_in[5];
  const float* Wo     = (const float*)d_in[6];
  const float* bo     = (const float*)d_in[7];
  float* out = (float*)d_out;

  // Workspace: 4 f16 tensors of NPROB*512*8 == B*N*T*64 elements each (~12.6MB)
  char* ws = (char*)d_ws;
  const size_t TSZ = (size_t)NPROB * Nn * 8 * sizeof(_Float16);  // 3,145,728 B
  _Float16* vp = (_Float16*)(ws);
  _Float16* kp = (_Float16*)(ws + TSZ);
  _Float16* qp = (_Float16*)(ws + 2 * TSZ);
  _Float16* o  = (_Float16*)(ws + 3 * TSZ);

  // 1) projections: 196608 threads == 768 * 256 exactly
  proj_kernel<<<768, 256, 0, stream>>>(values, keys, query, Wv, Wk, Wq,
                                       vp, kp, qp);
  // 2) attention: 384 problems x 32 query tiles, one wave32 each
  attn_kernel<<<NPROB * QTILES, 32, 0, stream>>>(qp, kp, vp, o);
  // 3) output projection: 1536 row tiles x 4 col tiles
  outproj_kernel<<<(Bb * Nn * Tt / 16) * 4, 32, 0, stream>>>(o, Wo, bo, out);
}
